// AttentionModule_32091995636361
// MI455X (gfx1250) — matrix operand order repaired and checked by reading, not compiled, not run on hardware
//
#include <hip/hip_runtime.h>
#include <math.h>

#define BB 8
#define CC 256
#define LL 2048
#define DD 32
#define KCHUNK 256
#define NOC (DD + DD + CC)          // 320 projection output channels

typedef __attribute__((ext_vector_type(16))) __bf16 v16bf;
typedef __attribute__((ext_vector_type(8)))  __bf16 v8bf;
typedef __attribute__((ext_vector_type(8)))  float  v8f;

// ---------------------------------------------------------------------------
// Kernel 0a: LDS-tiled transpose x(B,C,L) f32 -> Xt(B,L,C) bf16.
// Makes K-dim (channels) contiguous for WMMA B-operand loads.
// ---------------------------------------------------------------------------
#define TS 32
__global__ __launch_bounds__(256)
void pack_x_kernel(const float* __restrict__ x, __bf16* __restrict__ Xt)
{
    __shared__ float tile[TS][TS + 1];
    int lt = blockIdx.x % (LL / TS);
    int ct = (blockIdx.x / (LL / TS)) % (CC / TS);
    int b  = blockIdx.x / ((LL / TS) * (CC / TS));
    int tx = threadIdx.x & 31;
    int ty = threadIdx.x >> 5;                    // 0..7
    int l0 = lt * TS, c0 = ct * TS;

    #pragma unroll
    for (int p = 0; p < TS; p += 8)               // coalesced over l
        tile[ty + p][tx] = x[((size_t)b * CC + c0 + ty + p) * LL + l0 + tx];
    __syncthreads();
    #pragma unroll
    for (int p = 0; p < TS; p += 8)               // coalesced over c
        Xt[((size_t)b * LL + l0 + ty + p) * CC + c0 + tx] = (__bf16)tile[tx][ty + p];
}

// ---------------------------------------------------------------------------
// Kernel 0b: pack Wq|Wk|Wv -> Wall(320,256) bf16, biases -> biasf(320) f32.
// ---------------------------------------------------------------------------
__global__ __launch_bounds__(256)
void pack_w_kernel(const float* __restrict__ Wq, const float* __restrict__ bq,
                   const float* __restrict__ Wk, const float* __restrict__ bk,
                   const float* __restrict__ Wv, const float* __restrict__ bv,
                   __bf16* __restrict__ Wall, float* __restrict__ biasf)
{
    int o = blockIdx.x;                           // 0..319
    int c = threadIdx.x;                          // 0..255
    const float* W;
    const float* bias;
    int oo;
    if (o < DD)          { W = Wq; bias = bq; oo = o;          }
    else if (o < 2 * DD) { W = Wk; bias = bk; oo = o - DD;     }
    else                 { W = Wv; bias = bv; oo = o - 2 * DD; }
    Wall[(size_t)o * CC + c] = (__bf16)W[(size_t)oo * CC + c];
    if (c == 0) biasf[o] = bias[oo];
}

// ---------------------------------------------------------------------------
// Kernel 1: WMMA projections. Out(o,l) = sum_c Wall[o,c]*Xt[l,c] + bias[o].
// Block = 8 waves covering 2 o-tiles x 4 l-tiles; each wave owns one 16x16
// tile, K=256 folded as 8 x (16x16x32 bf16) WMMAs. Epilogue scatters into
// Qbf(B,L,D), Kbf(B,L,D), Vbf(B,C,L).
// ---------------------------------------------------------------------------
__global__ __launch_bounds__(256)
void proj_wmma_kernel(const __bf16* __restrict__ Xt,
                      const __bf16* __restrict__ Wall,
                      const float* __restrict__ biasf,
                      __bf16* __restrict__ Qbf, __bf16* __restrict__ Kbf,
                      __bf16* __restrict__ Vbf)
{
    const int lane = threadIdx.x & 31;
    const int wave = threadIdx.x >> 5;
    const int hi   = lane >> 4;
    const int ln   = lane & 15;

    const int NOT = NOC / 16;                     // 20 o-tiles
    const int LT  = LL / 16;                      // 128 l-tiles
    int ltb = blockIdx.x % (LT / 4);
    int otb = (blockIdx.x / (LT / 4)) % (NOT / 2);
    int b   = blockIdx.x / ((LT / 4) * (NOT / 2));

    int otile = otb * 2 + (wave >> 2);
    int ltile = ltb * 4 + (wave & 3);

    const __bf16* wbase = Wall + (size_t)(otile * 16 + ln) * CC;          // A row
    const __bf16* xbase = Xt + ((size_t)b * LL + ltile * 16 + ln) * CC + hi * 16; // B col

    v8f acc = {0.f, 0.f, 0.f, 0.f, 0.f, 0.f, 0.f, 0.f};
    #pragma unroll
    for (int kc = 0; kc < CC / 32; ++kc) {
        v8bf alo = *(const v8bf*)(wbase + kc * 32 + hi * 16);
        v8bf ahi = *(const v8bf*)(wbase + kc * 32 + hi * 16 + 8);
        v16bf aA = __builtin_shufflevector(alo, ahi, 0, 1, 2, 3, 4, 5, 6, 7,
                                                     8, 9, 10, 11, 12, 13, 14, 15);
        v16bf xB = *(const v16bf*)(xbase + kc * 32);                      // 16 contig ch
        acc = __builtin_amdgcn_wmma_f32_16x16x32_bf16(false, aA, false, xB,
                                                      (short)0, acc, false, false);
    }

    #pragma unroll
    for (int j = 0; j < 8; ++j) {                 // D: M=j+8*hi, N=ln
        int o = otile * 16 + j + 8 * hi;
        int l = ltile * 16 + ln;
        __bf16 h = (__bf16)(acc[j] + biasf[o]);
        if (o < DD)          Qbf[((size_t)b * LL + l) * DD + o] = h;
        else if (o < 2 * DD) Kbf[((size_t)b * LL + l) * DD + (o - DD)] = h;
        else                 Vbf[((size_t)b * CC + (o - 2 * DD)) * LL + l] = h;
    }
}

// ---------------------------------------------------------------------------
// Kernel 2: flash attention. One block = one (batch, 16-query tile).
// 8 waves x 32 lanes. bf16 WMMA (16x16x32, f32 accum) for S and O GEMMs.
// ---------------------------------------------------------------------------
__global__ __launch_bounds__(256)
void attn_kernel(const float* __restrict__ x, const float* __restrict__ gamma,
                 const __bf16* __restrict__ Qbf, const __bf16* __restrict__ Kbf,
                 const __bf16* __restrict__ Vbf, float* __restrict__ out)
{
    __shared__ float  S[16][KCHUNK];       // 16 KB: raw scores for this chunk
    __shared__ __bf16 P[16][KCHUNK];       //  8 KB: exp'd probabilities (bf16)
    __shared__ float  red[16][16];         //  1 KB: row reductions
    __shared__ float  m_run[16], l_run[16], alpha[16];

    const int tid  = threadIdx.x;
    const int lane = tid & 31;
    const int wave = tid >> 5;
    const int hi   = lane >> 4;            // lane half (WMMA operand layouts)
    const int ln   = lane & 15;

    const int mt = blockIdx.x % (LL / 16);
    const int b  = blockIdx.x / (LL / 16);
    const int m0 = mt * 16;

    // ---- Q A-tile (16x32 bf16), persistent. ----
    const __bf16* qbase = Qbf + ((size_t)b * LL + (m0 + ln)) * DD;
    v8bf qlo = *(const v8bf*)(qbase + hi * 16);
    v8bf qhi = *(const v8bf*)(qbase + hi * 16 + 8);
    v16bf qA = __builtin_shufflevector(qlo, qhi, 0, 1, 2, 3, 4, 5, 6, 7,
                                                 8, 9, 10, 11, 12, 13, 14, 15);

    if (tid < 16) { m_run[tid] = -INFINITY; l_run[tid] = 0.0f; }
    __syncthreads();

    v8f accO0 = {0.f, 0.f, 0.f, 0.f, 0.f, 0.f, 0.f, 0.f};
    v8f accO1 = {0.f, 0.f, 0.f, 0.f, 0.f, 0.f, 0.f, 0.f};

    const int row = tid >> 4;              // softmax phase: 16 rows x 16 threads
    const int sub = tid & 15;

    for (int k0 = 0; k0 < LL; k0 += KCHUNK) {
        // ---- phase 1: S-tiles. D=32 == WMMA K: one WMMA per 16x16 tile. ----
        #pragma unroll
        for (int t = 0; t < 2; ++t) {
            int ktile = wave * 2 + t;                          // 0..15
            int key   = k0 + ktile * 16 + ln;                  // B col = key
            const __bf16* kbase = Kbf + ((size_t)b * LL + key) * DD + hi * 16;
            v16bf kB = *(const v16bf*)kbase;
            v8f s = {0.f, 0.f, 0.f, 0.f, 0.f, 0.f, 0.f, 0.f};
            s = __builtin_amdgcn_wmma_f32_16x16x32_bf16(false, qA, false, kB,
                                                        (short)0, s, false, false);
            #pragma unroll
            for (int j = 0; j < 8; ++j)                        // D: M=j+8*hi, N=ln
                S[j + 8 * hi][ktile * 16 + ln] = s[j];
        }
        // prefetch this V chunk while scores are reduced
        {
            int ch = (wave * 2) * 16 + ln;
            const __bf16* vpf = Vbf + ((size_t)b * CC + ch) * LL + k0 + hi * 16;
            __builtin_prefetch(vpf, 0, 1);
        }
        __syncthreads();

        // ---- phase 2: online softmax over the chunk ----
        float lmax = -INFINITY;
        #pragma unroll
        for (int i = 0; i < KCHUNK / 16; ++i)
            lmax = fmaxf(lmax, S[row][sub + 16 * i]);
        red[row][sub] = lmax;
        __syncthreads();
        if (sub == 0) {
            float cm = red[row][0];
            #pragma unroll
            for (int i = 1; i < 16; ++i) cm = fmaxf(cm, red[row][i]);
            float nm = fmaxf(m_run[row], cm);
            alpha[row] = __expf(m_run[row] - nm);   // 0 on first chunk (-inf)
            m_run[row] = nm;
        }
        __syncthreads();
        float nm = m_run[row];
        float lsum = 0.0f;
        #pragma unroll
        for (int i = 0; i < KCHUNK / 16; ++i) {
            float p = __expf(S[row][sub + 16 * i] - nm);
            lsum += p;
            P[row][sub + 16 * i] = (__bf16)p;
        }
        red[row][sub] = lsum;
        __syncthreads();
        if (sub == 0) {
            float s16 = 0.0f;
            #pragma unroll
            for (int i = 0; i < 16; ++i) s16 += red[row][i];
            l_run[row] = alpha[row] * l_run[row] + s16;
        }
        __syncthreads();

        // ---- phase 3: rescale running O, accumulate O += P @ V ----
        #pragma unroll
        for (int j = 0; j < 8; ++j) {
            float a = alpha[j + 8 * hi];
            accO0[j] *= a;
            accO1[j] *= a;
        }
        #pragma unroll
        for (int t = 0; t < 2; ++t) {
            int cg = wave * 2 + t;
            int ch = cg * 16 + ln;                  // B col = channel
            const __bf16* vbase = Vbf + ((size_t)b * CC + ch) * LL + k0 + hi * 16;
            v8f acc = (t == 0) ? accO0 : accO1;
            #pragma unroll
            for (int kk = 0; kk < KCHUNK / 32; ++kk) {
                const __bf16* prow = &P[ln][kk * 32];       // A row = query ln
                v8bf plo = *(const v8bf*)(prow + hi * 16);
                v8bf phi = *(const v8bf*)(prow + hi * 16 + 8);
                v16bf pA = __builtin_shufflevector(plo, phi,
                        0, 1, 2, 3, 4, 5, 6, 7, 8, 9, 10, 11, 12, 13, 14, 15);
                v16bf vB = *(const v16bf*)(vbase + kk * 32);
                acc = __builtin_amdgcn_wmma_f32_16x16x32_bf16(false, pA, false, vB,
                                                              (short)0, acc,
                                                              false, false);
            }
            if (t == 0) accO0 = acc; else accO1 = acc;
        }
        __syncthreads();   // protect P before next chunk rewrites it
    }

    // ---- epilogue: out = gamma * O / l + x ----
    float g = gamma[0];
    #pragma unroll
    for (int t = 0; t < 2; ++t) {
        int ch = (wave * 2 + t) * 16 + ln;
        v8f acc = (t == 0) ? accO0 : accO1;
        #pragma unroll
        for (int j = 0; j < 8; ++j) {
            int m = j + 8 * hi;
            size_t idx = ((size_t)b * CC + ch) * LL + (m0 + m);
            out[idx] = g * acc[j] / l_run[m] + x[idx];
        }
    }
}

// ---------------------------------------------------------------------------
extern "C" void kernel_launch(void* const* d_in, const int* in_sizes, int n_in,
                              void* d_out, int out_size, void* d_ws, size_t ws_size,
                              hipStream_t stream) {
    (void)in_sizes; (void)n_in; (void)out_size; (void)ws_size;
    const float* x     = (const float*)d_in[0];
    const float* Wq    = (const float*)d_in[1];
    const float* bq    = (const float*)d_in[2];
    const float* Wk    = (const float*)d_in[3];
    const float* bk    = (const float*)d_in[4];
    const float* Wv    = (const float*)d_in[5];
    const float* bv    = (const float*)d_in[6];
    const float* gamma = (const float*)d_in[7];

    // Workspace layout (bytes):
    //   Qbf 1 MB | Kbf 1 MB | Vbf 8 MB | Xt 8 MB | Wall 160 KB | biasf 1.25 KB
    char* ws = (char*)d_ws;
    __bf16* Qbf   = (__bf16*)(ws);
    __bf16* Kbf   = (__bf16*)(ws + (size_t)1 * 1048576);
    __bf16* Vbf   = (__bf16*)(ws + (size_t)2 * 1048576);
    __bf16* Xt    = (__bf16*)(ws + (size_t)10 * 1048576);
    __bf16* Wall  = (__bf16*)(ws + (size_t)18 * 1048576);
    float*  biasf = (float*) (ws + (size_t)18 * 1048576 + 262144);

    // 0a: transpose+pack x -> Xt(B,L,C) bf16
    pack_x_kernel<<<dim3(BB * (CC / TS) * (LL / TS)), 256, 0, stream>>>(x, Xt);
    // 0b: pack weights/biases
    pack_w_kernel<<<dim3(NOC), 256, 0, stream>>>(Wq, bq, Wk, bk, Wv, bv, Wall, biasf);
    // 1: WMMA projections -> Qbf/Kbf/Vbf
    proj_wmma_kernel<<<dim3(BB * ((NOC / 16) / 2) * ((LL / 16) / 4)), 256, 0, stream>>>(
        Xt, Wall, biasf, Qbf, Kbf, Vbf);
    // 2: flash attention
    attn_kernel<<<dim3(BB * (LL / 16)), 256, 0, stream>>>(x, gamma, Qbf, Kbf, Vbf,
                                                          (float*)d_out);
}
